// ContrastLoss_61976378081883
// MI455X (gfx1250) — matrix operand order repaired and checked by reading, not compiled, not run on hardware
//
#include <hip/hip_runtime.h>
#include <stdint.h>

// MI455X / gfx1250, wave32. N=8192, H=512.
// fx_sim never materialized: exp + gat-weight + row-sum fused into GEMM epilogue.
// bf16x3 split GEMM on v_wmma_f32_16x16x32_bf16; tiles streamed with
// global_load_async_to_lds_b128 (ASYNCcnt) into double-buffered LDS.

typedef __bf16 v16bf __attribute__((ext_vector_type(16)));
typedef float  v8f   __attribute__((ext_vector_type(8)));

#define N_ROWS 8192
#define HDIM   512
#define BM     128
#define BN     128
#define BK     32
#define LDT    40                    // padded LDS row stride (bf16 elems)
#define NBLK   (N_ROWS / BN)         // 64 column blocks
#define TILE_B (BM * LDT * 2)        // 10240 bytes per operand tile
#define BUF_B  (4 * TILE_B)          // 40960 bytes per k-slab buffer (Ah,Al,Bh,Bl)
#define SMEM_B (2 * BUF_B)           // 81920 bytes double-buffered

__device__ __forceinline__ unsigned short f2bf_rne(float f) {
  unsigned u = __float_as_uint(f);
  u += 0x7FFFu + ((u >> 16) & 1u);   // round-to-nearest-even to bf16
  return (unsigned short)(u >> 16);
}
__device__ __forceinline__ float bf2f(unsigned short h) {
  return __uint_as_float(((unsigned)h) << 16);
}

// Async memory->LDS 16B transfer per lane (GVS mode: SGPR base + VGPR offset).
__device__ __forceinline__ void async_b128(unsigned lds_addr, unsigned voff,
                                           const void* sbase) {
  asm volatile("global_load_async_to_lds_b128 %0, %1, %2"
               :: "v"(lds_addr), "v"(voff), "s"(sbase) : "memory");
}

// ---------------------------------------------------------------------------
// Kernel 1: per-row L2 norms, nts = exp(<pro,x1>/tau), bf16 hi/lo split.
// ---------------------------------------------------------------------------
__global__ __launch_bounds__(256) void prep_split_kernel(
    const float* __restrict__ x1, const float* __restrict__ x2,
    const float* __restrict__ pro,
    unsigned short* __restrict__ zh1, unsigned short* __restrict__ zl1,
    unsigned short* __restrict__ zh2, unsigned short* __restrict__ zl2,
    float* __restrict__ nts) {
  const int wave = threadIdx.x >> 5;
  const int lane = threadIdx.x & 31;
  const int row  = blockIdx.x * 8 + wave;
  const size_t base = (size_t)row * HDIM + (size_t)lane * 16;

  float v[16];
  float ss = 0.f, dt = 0.f;
#pragma unroll
  for (int i = 0; i < 16; i += 4) {
    float4 a = *(const float4*)(x1 + base + i);
    float4 p = *(const float4*)(pro + base + i);
    v[i+0] = a.x; v[i+1] = a.y; v[i+2] = a.z; v[i+3] = a.w;
    ss += a.x*a.x + a.y*a.y + a.z*a.z + a.w*a.w;
    dt += a.x*p.x + a.y*p.y + a.z*p.z + a.w*p.w;
  }
#pragma unroll
  for (int m = 16; m >= 1; m >>= 1) {
    ss += __shfl_xor(ss, m, 32);
    dt += __shfl_xor(dt, m, 32);
  }
  float inv = 1.0f / fmaxf(sqrtf(ss), 1e-12f);
  if (lane == 0) nts[row] = __expf(dt * 2.0f);   // 1/tau = 2

  union Pack { unsigned short s[16]; uint4 q[2]; };
  Pack H, L;
#pragma unroll
  for (int i = 0; i < 16; ++i) {
    float z = v[i] * inv;
    unsigned short h = f2bf_rne(z);
    H.s[i] = h;
    L.s[i] = f2bf_rne(z - bf2f(h));
  }
  *(uint4*)(zh1 + base)     = H.q[0];  *(uint4*)(zh1 + base + 8) = H.q[1];
  *(uint4*)(zl1 + base)     = L.q[0];  *(uint4*)(zl1 + base + 8) = L.q[1];

  ss = 0.f;
#pragma unroll
  for (int i = 0; i < 16; i += 4) {
    float4 a = *(const float4*)(x2 + base + i);
    v[i+0] = a.x; v[i+1] = a.y; v[i+2] = a.z; v[i+3] = a.w;
    ss += a.x*a.x + a.y*a.y + a.z*a.z + a.w*a.w;
  }
#pragma unroll
  for (int m = 16; m >= 1; m >>= 1) ss += __shfl_xor(ss, m, 32);
  inv = 1.0f / fmaxf(sqrtf(ss), 1e-12f);
#pragma unroll
  for (int i = 0; i < 16; ++i) {
    float z = v[i] * inv;
    unsigned short h = f2bf_rne(z);
    H.s[i] = h;
    L.s[i] = f2bf_rne(z - bf2f(h));
  }
  *(uint4*)(zh2 + base)     = H.q[0];  *(uint4*)(zh2 + base + 8) = H.q[1];
  *(uint4*)(zl2 + base)     = L.q[0];  *(uint4*)(zl2 + base + 8) = L.q[1];
}

// Issue one 32-wide K-slab (4 tiles) into LDS buffer `buf` via async loads.
// Each of the 256 threads moves 32B per tile (2 x b128) => 8 async ops/thread.
__device__ __forceinline__ void stage_slab(
    unsigned smem, int buf, int rowT, int halfT, int bm, int bn, int kk,
    const void* zh1, const void* zl1, const void* zh2, const void* zl2) {
  const unsigned l  = smem + (unsigned)buf * BUF_B +
                      (unsigned)(rowT * LDT + halfT * 16) * 2u;
  const unsigned gA = (unsigned)((bm * BM + rowT) * HDIM + kk + halfT * 16) * 2u;
  const unsigned gB = (unsigned)((bn * BN + rowT) * HDIM + kk + halfT * 16) * 2u;
  async_b128(l,                  gA,      zh1);
  async_b128(l + 16,             gA + 16, zh1);
  async_b128(l +   TILE_B,       gA,      zl1);
  async_b128(l +   TILE_B + 16,  gA + 16, zl1);
  async_b128(l + 2*TILE_B,       gB,      zh2);
  async_b128(l + 2*TILE_B + 16,  gB + 16, zh2);
  async_b128(l + 3*TILE_B,       gB,      zl2);
  async_b128(l + 3*TILE_B + 16,  gB + 16, zl2);
}

// ---------------------------------------------------------------------------
// Kernel 2: 128x128 bf16x3 WMMA GEMM, async double-buffered LDS pipeline,
// fused exp/gat row-sum epilogue. Deterministic partials (one writer each).
// ---------------------------------------------------------------------------
__global__ __launch_bounds__(256) void simgat_gemm_kernel(
    const unsigned short* __restrict__ zh1, const unsigned short* __restrict__ zl1,
    const unsigned short* __restrict__ zh2, const unsigned short* __restrict__ zl2,
    const float* __restrict__ gat,
    float* __restrict__ Spart, float* __restrict__ Wpart) {
  extern __shared__ char smem_raw[];
  const unsigned smem = (unsigned)(uintptr_t)smem_raw;  // low 32b = LDS offset

  const int bn   = blockIdx.x;
  const int bm   = blockIdx.y;
  const int tid  = threadIdx.x;
  const int wave = tid >> 5;
  const int lane = tid & 31;
  const int lo   = lane & 15;
  const int hi   = lane >> 4;
  const int rowT  = tid >> 1;    // 0..127 staged tile row
  const int halfT = tid & 1;     // which 16-elem half of the 32-wide slab

  // Prefetch the 64 KB gat tile consumed by the epilogue (global_prefetch_b8).
  {
    const int r = tid >> 1;
    const int c = (tid & 1) * 64;
    __builtin_prefetch(gat + (size_t)(bm * BM + r) * N_ROWS + bn * BN + c, 0, 3);
  }

  v8f acc[8];
  const v8f vzero = {0.f, 0.f, 0.f, 0.f, 0.f, 0.f, 0.f, 0.f};
#pragma unroll
  for (int s = 0; s < 8; ++s) acc[s] = vzero;

  union Frag { uint4 q[2]; v16bf bf; };

  // Prologue: slabs 0 and 1 in flight (16 async ops outstanding per thread).
  stage_slab(smem, 0, rowT, halfT, bm, bn, 0,  zh1, zl1, zh2, zl2);
  stage_slab(smem, 1, rowT, halfT, bm, bn, BK, zh1, zl1, zh2, zl2);

  for (int kk = 0; kk < HDIM; kk += BK) {
    const int buf = (kk / BK) & 1;
    // Oldest 8 async ops == slab kk (in-order completion on ASYNCcnt).
    if (kk + BK < HDIM) asm volatile("s_wait_asynccnt 0x8" ::: "memory");
    else                asm volatile("s_wait_asynccnt 0x0" ::: "memory");
    __syncthreads();   // slab kk visible to all waves

    const unsigned short* Ah = (const unsigned short*)(smem_raw + buf * BUF_B);
    const unsigned short* Al = (const unsigned short*)(smem_raw + buf * BUF_B + TILE_B);
    const unsigned short* Bh = (const unsigned short*)(smem_raw + buf * BUF_B + 2 * TILE_B);
    const unsigned short* Bl = (const unsigned short*)(smem_raw + buf * BUF_B + 3 * TILE_B);

    // A fragment (16x32 bf16): lanes 0-15 K0..15, lanes 16-31 K16..31
    Frag aH, aL;
    const int aBase = (wave * 16 + lo) * LDT;
    aH.q[0] = *(const uint4*)&Ah[aBase + hi * 16];
    aH.q[1] = *(const uint4*)&Ah[aBase + 8 + hi * 16];
    aL.q[0] = *(const uint4*)&Al[aBase + hi * 16];
    aL.q[1] = *(const uint4*)&Al[aBase + 8 + hi * 16];

#pragma unroll
    for (int s = 0; s < 8; ++s) {
      // B fragment (32x16 bf16): lanes 0-15 K0..15, lanes 16-31 K16..31 (N = lo)
      Frag bH, bL;
      const int bBase = (s * 16 + lo) * LDT + hi * 16;
      bH.q[0] = *(const uint4*)&Bh[bBase];
      bH.q[1] = *(const uint4*)&Bh[bBase + 8];
      bL.q[0] = *(const uint4*)&Bl[bBase];
      bL.q[1] = *(const uint4*)&Bl[bBase + 8];

      acc[s] = __builtin_amdgcn_wmma_f32_16x16x32_bf16(
          false, aH.bf, false, bH.bf, (short)0, acc[s], false, false);
      acc[s] = __builtin_amdgcn_wmma_f32_16x16x32_bf16(
          false, aH.bf, false, bL.bf, (short)0, acc[s], false, false);
      acc[s] = __builtin_amdgcn_wmma_f32_16x16x32_bf16(
          false, aL.bf, false, bH.bf, (short)0, acc[s], false, false);
    }

    // All of this wave's LDS reads of buf complete, then sync so no wave is
    // still reading buf before we overwrite it with slab kk+2.
    asm volatile("s_wait_dscnt 0x0" ::: "memory");
    __syncthreads();
    if (kk + 2 * BK < HDIM)
      stage_slab(smem, buf, rowT, halfT, bm, bn, kk + 2 * BK, zh1, zl1, zh2, zl2);
  }

  // Fused epilogue: e = exp(c/tau); accumulate e and e*gat over this col block.
  // C/D layout: VGPR v -> (M = v + 8*hi, N = lo) within each 16x16 sub-tile.
  const int rowBase = bm * BM + wave * 16;
  float sS[8], sW[8];
#pragma unroll
  for (int v = 0; v < 8; ++v) { sS[v] = 0.f; sW[v] = 0.f; }

#pragma unroll
  for (int s = 0; s < 8; ++s) {
    const int col = bn * BN + s * 16 + lo;
#pragma unroll
    for (int v = 0; v < 8; ++v) {
      const int row = rowBase + v + 8 * hi;
      const float e = __expf(acc[s][v] * 2.0f);
      const float g = gat[(size_t)row * N_ROWS + col];
      sS[v] += e;
      sW[v] += e * g;
    }
  }

#pragma unroll
  for (int v = 0; v < 8; ++v) {
#pragma unroll
    for (int m = 8; m >= 1; m >>= 1) {   // reduce the 16 lanes sharing a row
      sS[v] += __shfl_xor(sS[v], m, 16);
      sW[v] += __shfl_xor(sW[v], m, 16);
    }
    if (lo == 0) {
      const int row = rowBase + v + 8 * hi;
      Spart[(size_t)row * NBLK + bn] = sS[v];
      Wpart[(size_t)row * NBLK + bn] = sW[v];
    }
  }
}

// ---------------------------------------------------------------------------
// Kernel 3: reduce 64 partials/row, -log((nts+W)/(nts+S+1e-8)), mean.
// ---------------------------------------------------------------------------
__global__ __launch_bounds__(1024) void finalize_kernel(
    const float* __restrict__ nts, const float* __restrict__ Spart,
    const float* __restrict__ Wpart, float* __restrict__ out) {
  __shared__ float red[1024];
  float local = 0.f;
  for (int r = threadIdx.x; r < N_ROWS; r += 1024) {
    float S = 0.f, W = 0.f;
    for (int b = 0; b < NBLK; ++b) {
      S += Spart[(size_t)r * NBLK + b];
      W += Wpart[(size_t)r * NBLK + b];
    }
    const float n = nts[r];
    const float fenzi = n + W;
    const float fenmu = n + S + 1e-8f;
    local += -logf(fenzi / fenmu);
  }
  red[threadIdx.x] = local;
  __syncthreads();
  for (int off = 512; off > 0; off >>= 1) {
    if (threadIdx.x < off) red[threadIdx.x] += red[threadIdx.x + off];
    __syncthreads();
  }
  if (threadIdx.x == 0) out[0] = red[0] / (float)N_ROWS;
}

// ---------------------------------------------------------------------------
extern "C" void kernel_launch(void* const* d_in, const int* in_sizes, int n_in,
                              void* d_out, int out_size, void* d_ws, size_t ws_size,
                              hipStream_t stream) {
  const float* x1  = (const float*)d_in[0];
  const float* x2  = (const float*)d_in[1];
  const float* pro = (const float*)d_in[2];
  const float* gat = (const float*)d_in[3];
  float* out = (float*)d_out;

  // Workspace (~36.1 MiB): 4 x 8 MiB bf16 split operands, nts 32 KiB,
  // Spart/Wpart 2 x 2 MiB deterministic per-(row, colBlock) partials.
  char* ws = (char*)d_ws;
  const size_t zbytes = (size_t)N_ROWS * HDIM * sizeof(unsigned short);
  unsigned short* zh1 = (unsigned short*)(ws);
  unsigned short* zl1 = (unsigned short*)(ws + zbytes);
  unsigned short* zh2 = (unsigned short*)(ws + 2 * zbytes);
  unsigned short* zl2 = (unsigned short*)(ws + 3 * zbytes);
  float* nts   = (float*)(ws + 4 * zbytes);
  float* Spart = (float*)(ws + 4 * zbytes + (size_t)N_ROWS * sizeof(float));
  float* Wpart = Spart + (size_t)N_ROWS * NBLK;

  prep_split_kernel<<<N_ROWS / 8, 256, 0, stream>>>(x1, x2, pro,
                                                    zh1, zl1, zh2, zl2, nts);
  simgat_gemm_kernel<<<dim3(NBLK, N_ROWS / BM), 256, SMEM_B, stream>>>(
      zh1, zl1, zh2, zl2, gat, Spart, Wpart);
  finalize_kernel<<<1, 1024, 0, stream>>>(nts, Spart, Wpart, out);
}
